// NNUEModel_46497315946985
// MI455X (gfx1250) — matrix operand order repaired and checked by reading, not compiled, not run on hardware
//
#include <hip/hip_runtime.h>

// ---------------------------------------------------------------------------
// NNUE forward pass for MI455X (gfx1250, wave32).
//   d_in[0] white_features [4096,40960] f32   d_in[5] fc1_b [32]   f32
//   d_in[1] black_features [4096,40960] f32   d_in[6] fc2_w [32,32] f32
//   d_in[2] ft_w [256,40960] f32              d_in[7] fc2_b [32]   f32
//   d_in[3] ft_b [256] f32                    d_in[8] fc3_w [1,32] f32
//   d_in[4] fc1_w [32,512] f32                d_in[9] fc3_b [1]    f32
//   d_out: [4096] f32
//
// Strategy: exploit ~30/40960 sparsity of the feature rows. The 1.34 GB
// feature scan is the HBM floor (~58us at 23.3 TB/s) and is done with
// NON-TEMPORAL b128 loads so the one-shot stream does not evict the
// L2-resident 21 MB f16 transpose of ft_w that the gather phase hits.
// The 512->32->32->1 MLP tail runs on v_wmma_f32_16x16x32_f16.
// ---------------------------------------------------------------------------

typedef _Float16 v16h __attribute__((ext_vector_type(16)));
typedef _Float16 v8h  __attribute__((ext_vector_type(8)));
typedef float    v8f  __attribute__((ext_vector_type(8)));
typedef float    v4f  __attribute__((ext_vector_type(4)));

#define FEAT   40960
#define NROWS  4096
#define H1     256
#define MAXNNZ 64

// ---------------------------------------------------------------------------
// Kernel A: LDS-tiled transpose + f32->f16 convert of ft_w [256,40960]
//           into ft_wT [40960,256] so a gathered feature is 512 contiguous B.
// ---------------------------------------------------------------------------
__global__ void __launch_bounds__(1024)
k_transpose_ftw(const float* __restrict__ ft_w, _Float16* __restrict__ ft_wT) {
    __shared__ float tile[32][33];            // +1 pad: no bank conflicts
    const int f0 = blockIdx.x * 32;
    const int h0 = blockIdx.y * 32;
    const int tx = threadIdx.x, ty = threadIdx.y;
    tile[ty][tx] = ft_w[(size_t)(h0 + ty) * FEAT + (f0 + tx)];   // coalesced in f
    __syncthreads();
    ft_wT[(size_t)(f0 + ty) * H1 + (h0 + tx)] = (_Float16)tile[tx][ty]; // coalesced in h
}

// ---------------------------------------------------------------------------
// Kernel A2: convert the tiny fc1/fc2 weights to f16 (WMMA operand format).
// ---------------------------------------------------------------------------
__global__ void k_convert_w(const float* __restrict__ fc1_w,
                            const float* __restrict__ fc2_w,
                            _Float16* __restrict__ fc1_wh,
                            _Float16* __restrict__ fc2_wh) {
    const int i = blockIdx.x * blockDim.x + threadIdx.x;
    if (i < 32 * 512) fc1_wh[i] = (_Float16)fc1_w[i];
    if (i < 32 * 32)  fc2_wh[i] = (_Float16)fc2_w[i];
}

// ---------------------------------------------------------------------------
// Kernel B: per (batch row, side): stream the 160KB feature row with
// NON-TEMPORAL float4 loads (read-once data; protect L2-resident ft_wT),
// record nonzero indices in LDS, then gather-accumulate the corresponding
// ft_wT rows (L2 hits), crelu, emit f16 into x_cat[4096,512].
// ---------------------------------------------------------------------------
__global__ void __launch_bounds__(256)
k_feature_transform(const float* __restrict__ wfeat,
                    const float* __restrict__ bfeat,
                    const _Float16* __restrict__ ft_wT,
                    const float* __restrict__ ft_b,
                    _Float16* __restrict__ x_cat) {
    const int row  = blockIdx.x;     // 0..4095
    const int side = blockIdx.y;     // 0 = white, 1 = black
    const float* feat = (side == 0 ? wfeat : bfeat) + (size_t)row * FEAT;

    __shared__ int s_idx[MAXNNZ];
    __shared__ int s_cnt;
    const int tid = threadIdx.x;
    if (tid == 0) s_cnt = 0;
    __syncthreads();

    const v4f* f4 = (const v4f*)feat;             // FEAT/4 = 10240 vectors
    for (int i = tid; i < FEAT / 4; i += 256) {
        v4f v = __builtin_nontemporal_load(f4 + i);   // b128, th:NT
        if (v.x != 0.f || v.y != 0.f || v.z != 0.f || v.w != 0.f) {
            if (v.x != 0.f) { int p = atomicAdd(&s_cnt, 1); if (p < MAXNNZ) s_idx[p] = 4 * i + 0; }
            if (v.y != 0.f) { int p = atomicAdd(&s_cnt, 1); if (p < MAXNNZ) s_idx[p] = 4 * i + 1; }
            if (v.z != 0.f) { int p = atomicAdd(&s_cnt, 1); if (p < MAXNNZ) s_idx[p] = 4 * i + 2; }
            if (v.w != 0.f) { int p = atomicAdd(&s_cnt, 1); if (p < MAXNNZ) s_idx[p] = 4 * i + 3; }
        }
    }
    __syncthreads();

    int n = s_cnt; if (n > MAXNNZ) n = MAXNNZ;
    float acc = ft_b[tid];                        // thread owns channel h = tid
    for (int j = 0; j < n; ++j)
        acc += (float)ft_wT[(size_t)s_idx[j] * H1 + tid];   // L2-resident gather
    acc = fminf(fmaxf(acc, 0.f), 1.f);            // crelu
    x_cat[(size_t)row * 512 + side * 256 + tid] = (_Float16)acc;
}

// ---------------------------------------------------------------------------
// Kernel C: one wave (32 lanes) per 16-row batch tile.
//   fc1: [16,512] x [512,32] via 32x v_wmma_f32_16x16x32_f16 (bias in C)
//   fc2: [16,32]  x [32,32]  via 2x  WMMA (A from LDS)
//   fc3: [16,32]  x [32,1]   via 16-lane VALU dot product
// Fragment layouts used here (contiguous k, same order in A and B):
//   A (16x32 f16): lane m=L&15, koff=16*(L>>4): halves K=koff..koff+7 then
//                  K=koff+8..koff+15 (two 16B loads).
//   B (32x16 f16): lane n=L&15: halves K=16*(L>>4)..+15 (one 32B load of
//                  B^T row n, i.e. the weight matrix row itself).
//   C/D (16x16 f32): lane L, vgpr r -> (m = r + 8*(L>>4), n = L&15).
// ---------------------------------------------------------------------------
__global__ void __launch_bounds__(32)
k_fc_wmma(const _Float16* __restrict__ x_cat,
          const _Float16* __restrict__ fc1_wh, const float* __restrict__ fc1_b,
          const _Float16* __restrict__ fc2_wh, const float* __restrict__ fc2_b,
          const float* __restrict__ fc3_w,     const float* __restrict__ fc3_b,
          float* __restrict__ out) {
    const int m0   = blockIdx.x * 16;
    const int lane = threadIdx.x;
    const int lm   = lane & 15;
    const int lh   = lane >> 4;        // 0/1 half-wave

    __shared__ _Float16 s_x[16 * 32];
    __shared__ _Float16 s_y[16 * 32];

    // ---- fc1 ----
    v8f acc[2];
    for (int nt = 0; nt < 2; ++nt) {
        const float bv = fc1_b[nt * 16 + lm];    // bias depends on column n only
        for (int r = 0; r < 8; ++r) acc[nt][r] = bv;
    }
    const _Float16* arow = x_cat + (size_t)(m0 + lm) * 512;
    for (int k0 = 0; k0 < 512; k0 += 32) {
        union { v16h v; v8h h[2]; } a;
        a.h[0] = *(const v8h*)(arow + k0 + 16 * lh);
        a.h[1] = *(const v8h*)(arow + k0 + 16 * lh + 8);
        for (int nt = 0; nt < 2; ++nt) {
            const v16h b = *(const v16h*)(fc1_wh + (size_t)(nt * 16 + lm) * 512 + k0 + 16 * lh);
            acc[nt] = __builtin_amdgcn_wmma_f32_16x16x32_f16(
                false, a.v, false, b, (short)0, acc[nt], false, false);
        }
    }
    // crelu -> LDS [16 m][32 n] f16
    for (int nt = 0; nt < 2; ++nt) {
        const int n = nt * 16 + lm;
        for (int r = 0; r < 8; ++r) {
            const int m = r + 8 * lh;
            float v = fminf(fmaxf(acc[nt][r], 0.f), 1.f);
            s_x[m * 32 + n] = (_Float16)v;
        }
    }
    __syncthreads();

    // ---- fc2 ----
    v8f acc2[2];
    for (int nt = 0; nt < 2; ++nt) {
        const float bv = fc2_b[nt * 16 + lm];
        for (int r = 0; r < 8; ++r) acc2[nt][r] = bv;
    }
    union { v16h v; v8h h[2]; } a2;
    const _Float16* a2row = s_x + lm * 32;
    a2.h[0] = *(const v8h*)(a2row + 16 * lh);
    a2.h[1] = *(const v8h*)(a2row + 16 * lh + 8);
    for (int nt = 0; nt < 2; ++nt) {
        const v16h b2 = *(const v16h*)(fc2_wh + (nt * 16 + lm) * 32 + 16 * lh);
        acc2[nt] = __builtin_amdgcn_wmma_f32_16x16x32_f16(
            false, a2.v, false, b2, (short)0, acc2[nt], false, false);
    }
    for (int nt = 0; nt < 2; ++nt) {
        const int n = nt * 16 + lm;
        for (int r = 0; r < 8; ++r) {
            const int m = r + 8 * lh;
            float v = fminf(fmaxf(acc2[nt][r], 0.f), 1.f);
            s_y[m * 32 + n] = (_Float16)v;
        }
    }
    __syncthreads();

    // ---- fc3 ----
    if (lane < 16) {
        float accv = fc3_b[0];
        for (int j = 0; j < 32; ++j)
            accv += (float)s_y[lane * 32 + j] * fc3_w[j];
        out[m0 + lane] = accv;
    }
}

// ---------------------------------------------------------------------------
extern "C" void kernel_launch(void* const* d_in, const int* in_sizes, int n_in,
                              void* d_out, int out_size, void* d_ws, size_t ws_size,
                              hipStream_t stream) {
    (void)in_sizes; (void)n_in; (void)out_size; (void)ws_size;
    const float* wfeat = (const float*)d_in[0];
    const float* bfeat = (const float*)d_in[1];
    const float* ft_w  = (const float*)d_in[2];
    const float* ft_b  = (const float*)d_in[3];
    const float* fc1_w = (const float*)d_in[4];
    const float* fc1_b = (const float*)d_in[5];
    const float* fc2_w = (const float*)d_in[6];
    const float* fc2_b = (const float*)d_in[7];
    const float* fc3_w = (const float*)d_in[8];
    const float* fc3_b = (const float*)d_in[9];
    float* out = (float*)d_out;

    // workspace layout (bytes)
    char* ws = (char*)d_ws;
    _Float16* ft_wT  = (_Float16*)(ws);                       // 40960*256*2 = 20,971,520
    _Float16* x_cat  = (_Float16*)(ws + 20971520);            //  4096*512*2 =  4,194,304
    _Float16* fc1_wh = (_Float16*)(ws + 20971520 + 4194304);  //    32*512*2 =     32,768
    _Float16* fc2_wh = (_Float16*)(ws + 20971520 + 4194304 + 32768); // 32*32*2 = 2,048

    k_transpose_ftw<<<dim3(FEAT / 32, H1 / 32), dim3(32, 32), 0, stream>>>(ft_w, ft_wT);
    k_convert_w<<<64, 256, 0, stream>>>(fc1_w, fc2_w, fc1_wh, fc2_wh);
    k_feature_transform<<<dim3(NROWS, 2), 256, 0, stream>>>(wfeat, bfeat, ft_wT, ft_b, x_cat);
    k_fc_wmma<<<NROWS / 16, 32, 0, stream>>>(x_cat, fc1_wh, fc1_b, fc2_wh, fc2_b,
                                             fc3_w, fc3_b, out);
}
